// TemporalModule_37898791420550
// MI455X (gfx1250) — matrix operand order repaired and checked by reading, not compiled, not run on hardware
//
#include <hip/hip_runtime.h>
#include <hip/hip_bf16.h>

#define C_DIM 320
#define P_PAST 32
#define R_SEL 8
#define H_HEADS 8
#define DH 40
#define TOKENS_PER_BLOCK 16
#define HALF_TOKENS 8
#define NTHREADS 256
#define W_ELEMS (C_DIM * C_DIM) /* 102400 */

typedef __attribute__((ext_vector_type(16))) __bf16 v16bf;
typedef __attribute__((ext_vector_type(8)))  float  v8f;
typedef __attribute__((ext_vector_type(8)))  unsigned short ush8;
typedef __attribute__((ext_vector_type(16))) unsigned short ush16;
typedef __attribute__((ext_vector_type(4)))  float  f4;

__device__ __forceinline__ unsigned short f2bf(float f) {
  unsigned u = __builtin_bit_cast(unsigned, f);
  u += 0x7FFFu + ((u >> 16) & 1u);            // round-to-nearest-even
  return (unsigned short)(u >> 16);
}
__device__ __forceinline__ float bf2f(unsigned short b) {
  unsigned u = ((unsigned)b) << 16;
  return __builtin_bit_cast(float, u);
}

// A-fragment (16x32 bf16, M x K): row = lane&15.
// elems 0..7  = K kt*32 + kb + 0..7   (kb = 0 lanes 0-15, 8 lanes 16-31)
// elems 8..15 = K kt*32 + kb + 16..23
__device__ __forceinline__ v16bf load_a_frag(const unsigned short* row, int kt, int lane) {
  int kb = kt * 32 + ((lane & 16) ? 8 : 0);
  ush8 lo = *(const ush8*)(row + kb);
  ush8 hi = *(const ush8*)(row + kb + 16);
  ush16 u;
#pragma unroll
  for (int i = 0; i < 8; ++i) { u[i] = lo[i]; u[8 + i] = hi[i]; }
  return __builtin_bit_cast(v16bf, u);
}

// B-fragment (32x16 bf16, K x N) from a K-contiguous weight image wb[n*C + k]:
// col n = lane&15; elems 0..7 = K kt*32 + kb + 0..7, elems 8..15 = +16..23 (kb = 0 / 8).
__device__ __forceinline__ v16bf load_b_frag(const unsigned short* wb, int n, int kt, int lane) {
  const unsigned short* p = wb + (size_t)n * C_DIM + kt * 32 + ((lane & 16) ? 8 : 0);
  ush8 lo = *(const ush8*)(p);
  ush8 hi = *(const ush8*)(p + 16);
  ush16 u;
#pragma unroll
  for (int i = 0; i < 8; ++i) { u[i] = lo[i]; u[8 + i] = hi[i]; }
  return __builtin_bit_cast(v16bf, u);
}

__device__ __forceinline__ v8f wmma_bf16(v16bf a, v16bf b, v8f c) {
  return __builtin_amdgcn_wmma_f32_16x16x32_bf16(false, a, false, b, (short)0, c, false, false);
}

// ws layout (bf16): [0] Wq^T  [1] Wk^T  [2] Wv^T  [3] Wo^T  [4] Wk row-major
__global__ void prep_weights_kernel(const float* __restrict__ Wq, const float* __restrict__ Wk,
                                    const float* __restrict__ Wv, const float* __restrict__ Wo,
                                    unsigned short* __restrict__ ws) {
  int e = blockIdx.x * blockDim.x + threadIdx.x;
  if (e >= W_ELEMS) return;
  int r = e / C_DIM, c = e % C_DIM;
  ws[0 * W_ELEMS + c * C_DIM + r] = f2bf(Wq[e]);
  ws[1 * W_ELEMS + c * C_DIM + r] = f2bf(Wk[e]);
  ws[2 * W_ELEMS + c * C_DIM + r] = f2bf(Wv[e]);
  ws[3 * W_ELEMS + c * C_DIM + r] = f2bf(Wo[e]);
  ws[4 * W_ELEMS + e]             = f2bf(Wk[e]);
}

__global__ __launch_bounds__(NTHREADS)
void temporal_attn_kernel(const float* __restrict__ hidden,
                          const float* __restrict__ cached,
                          const unsigned short* __restrict__ wbf,
                          const float* __restrict__ bo,
                          float* __restrict__ out) {
  extern __shared__ char smem[];
  unsigned short* sNow  = (unsigned short*)(smem);            // 16x320 bf16 (raw now rows)
  unsigned short* sQ    = (unsigned short*)(smem + 10240);    // 16x320 bf16
  unsigned short* sKnow = (unsigned short*)(smem + 20480);    // 16x320 bf16
  float*          sU    = (float*)(smem + 30720);             // 16x320 f32 (Wk@q)
  int*            sIdx  = (int*)(smem + 51200);               // 16x8 top-k indices
  float*          sProb = (float*)(smem + 51712);             // 8x8x12 f32
  unsigned short* sGat  = (unsigned short*)(smem + 54784);    // 64x320 bf16 gathered raw rows
  unsigned short* sK    = (unsigned short*)(smem + 95744);    // 72x320 bf16 (8 tok x 9 slots)
  unsigned short* sMix  = sK;                                 // 64x320 overlay (prob-mixed raw)
  unsigned short* sCtx  = (unsigned short*)(smem + 141824);   // 16x320 bf16
  // total 152064 bytes -> 2 workgroups per 320KB WGP

  const int tid  = threadIdx.x;
  const int lane = tid & 31;
  const int wave = tid >> 5;
  const int tok0 = blockIdx.x * TOKENS_PER_BLOCK;

  const unsigned short* wtq = wbf;
  const unsigned short* wtk = wbf + 1 * W_ELEMS;
  const unsigned short* wtv = wbf + 2 * W_ELEMS;
  const unsigned short* wto = wbf + 3 * W_ELEMS;
  const unsigned short* wkr = wbf + 4 * W_ELEMS;

  // Phase 1: stage 16 "now" rows as bf16
  for (int e = tid; e < TOKENS_PER_BLOCK * (C_DIM / 4); e += NTHREADS) {
    int t = e / 80, c4 = e % 80;
    f4 v = *(const f4*)(hidden + (size_t)(tok0 + t) * C_DIM + c4 * 4);
#pragma unroll
    for (int j = 0; j < 4; ++j) sNow[(size_t)t * C_DIM + c4 * 4 + j] = f2bf(v[j]);
  }
  __syncthreads();

  // Phase 2: Q = now@Wq, Know = now@Wk  (WMMA, 16x320 each)
  for (int task = wave; task < 40; task += 8) {
    int g = task / 20, ct = task % 20;
    const unsigned short* wb = g ? wtk : wtq;
    unsigned short* dst = g ? sKnow : sQ;
    const unsigned short* arow = sNow + (size_t)(lane & 15) * C_DIM;
    int n = ct * 16 + (lane & 15);
    v8f acc = {};
#pragma unroll
    for (int kt = 0; kt < 10; ++kt)
      acc = wmma_bf16(load_a_frag(arow, kt, lane), load_b_frag(wb, n, kt, lane), acc);
#pragma unroll
    for (int vi = 0; vi < 8; ++vi) {
      int m = (lane & 16) ? (8 + vi) : vi;
      dst[(size_t)m * C_DIM + n] = f2bf(acc[vi]);
    }
  }
  __syncthreads();

  // Phase 3: U = Q @ Wk^T   (identity: score = cached @ (Wk q))
  for (int ct = wave; ct < 20; ct += 8) {
    const unsigned short* arow = sQ + (size_t)(lane & 15) * C_DIM;
    int n = ct * 16 + (lane & 15);
    v8f acc = {};
#pragma unroll
    for (int kt = 0; kt < 10; ++kt)
      acc = wmma_bf16(load_a_frag(arow, kt, lane), load_b_frag(wkr, n, kt, lane), acc);
#pragma unroll
    for (int vi = 0; vi < 8; ++vi) {
      int m = (lane & 16) ? (8 + vi) : vi;
      sU[(size_t)m * C_DIM + n] = acc[vi];
    }
  }
  __syncthreads();

  // Phase 4: scores over all 32 past rows. This is the HBM-roofline phase
  // (671 MB streamed once): cooperative row-dot, lanes read CONTIGUOUS
  // float4 chunks (fully coalesced 512B/instr), shuffle-tree reduce; the
  // row score is parked in lane p so the top-8 shuffle select works as-is.
  for (int tt = 0; tt < 2; ++tt) {
    int t = wave * 2 + tt;
    const float* base = cached + (size_t)(tok0 + t) * P_PAST * C_DIM;
    const float* u = sU + (size_t)t * C_DIM;
    f4 u0 = *(const f4*)(u + lane * 4);
    f4 u1 = *(const f4*)(u + (lane + 32) * 4);
    f4 u2 = {0.f, 0.f, 0.f, 0.f};
    if (lane < 16) u2 = *(const f4*)(u + (lane + 64) * 4);
    float s = 0.f;                       // lane p will hold score of past-row p
#pragma unroll 4
    for (int p = 0; p < P_PAST; ++p) {
      const float* row = base + (size_t)p * C_DIM;
      __builtin_prefetch(row + 2 * C_DIM + lane * 16, 0, 0);  // global_prefetch_b8
      f4 a0 = *(const f4*)(row + lane * 4);
      f4 a1 = *(const f4*)(row + (lane + 32) * 4);
      float part = a0[0] * u0[0] + a0[1] * u0[1] + a0[2] * u0[2] + a0[3] * u0[3]
                 + a1[0] * u1[0] + a1[1] * u1[1] + a1[2] * u1[2] + a1[3] * u1[3];
      if (lane < 16) {
        f4 a2 = *(const f4*)(row + (lane + 64) * 4);
        part += a2[0] * u2[0] + a2[1] * u2[1] + a2[2] * u2[2] + a2[3] * u2[3];
      }
#pragma unroll
      for (int off = 16; off > 0; off >>= 1) part += __shfl_xor(part, off);
      if (lane == p) s = part;
    }
    for (int r = 0; r < R_SEL; ++r) {    // descending, ties -> lower index
      float cv = s; int ci = lane;
#pragma unroll
      for (int off = 16; off > 0; off >>= 1) {
        float ov = __shfl_xor(cv, off);
        int   oi = __shfl_xor(ci, off);
        if (ov > cv || (ov == cv && oi < ci)) { cv = ov; ci = oi; }
      }
      if (lane == 0) sIdx[t * R_SEL + r] = ci;
      if (lane == ci) s = -__builtin_inff();
    }
  }
  __syncthreads();

  const float inv_sqrt_dh = 0.15811388300841897f; // 1/sqrt(40)
  const float kfreq       = 0.08304820237218406f; // log2(1e4)/160

  for (int half = 0; half < 2; ++half) {
    const int tb = half * HALF_TOKENS;

    // 5a: gather 8 selected raw cached rows per token
    for (int e = tid; e < 64 * 80; e += NTHREADS) {
      int r = e / 80, c4 = e % 80;
      int tl = r >> 3, sel = r & 7;
      int prow = sIdx[(tb + tl) * R_SEL + sel];
      f4 v = *(const f4*)(cached + ((size_t)(tok0 + tb + tl) * P_PAST + prow) * C_DIM + c4 * 4);
#pragma unroll
      for (int j = 0; j < 4; ++j) sGat[(size_t)r * C_DIM + c4 * 4 + j] = f2bf(v[j]);
    }
    __syncthreads();

    // 5b: K_sel = gathered @ Wk (WMMA), scattered into [token][slot] layout
    for (int task = wave; task < 80; task += 8) {
      int rt = task / 20, ct = task % 20;
      const unsigned short* arow = sGat + (size_t)(rt * 16 + (lane & 15)) * C_DIM;
      int n = ct * 16 + (lane & 15);
      v8f acc = {};
#pragma unroll
      for (int kt = 0; kt < 10; ++kt)
        acc = wmma_bf16(load_a_frag(arow, kt, lane), load_b_frag(wtk, n, kt, lane), acc);
#pragma unroll
      for (int vi = 0; vi < 8; ++vi) {
        int m = (lane & 16) ? (8 + vi) : vi;
        int r = rt * 16 + m, tl = r >> 3, sl = r & 7;
        sK[((size_t)tl * 9 + sl) * C_DIM + n] = f2bf(acc[vi]);
      }
    }
    __syncthreads();

    // 5c: attention logits with RoPE fused at read (key position = slot 0..8; q pos 0 = identity)
    for (int e = tid; e < HALF_TOKENS * H_HEADS * 9; e += NTHREADS) {
      int t = e / 72, h = (e / 9) % 8, s = e % 9;
      const unsigned short* qp = sQ + (size_t)(tb + t) * C_DIM + h * DH;
      const unsigned short* kp = (s < 8) ? (sK + ((size_t)t * 9 + s) * C_DIM + h * DH)
                                         : (sKnow + (size_t)(tb + t) * C_DIM + h * DH);
      float acc = 0.f;
      for (int i2 = 0; i2 < 20; ++i2) {
        int fi = h * 20 + i2;
        float ang = (float)s * exp2f(-(float)fi * kfreq);
        float sn, cs;
        __sincosf(ang, &sn, &cs);
        float k0 = bf2f(kp[i2 * 2]), k1 = bf2f(kp[i2 * 2 + 1]);
        float q0 = bf2f(qp[i2 * 2]), q1 = bf2f(qp[i2 * 2 + 1]);
        acc += q0 * (k0 * cs - k1 * sn) + q1 * (k0 * sn + k1 * cs);
      }
      sProb[(t * 8 + h) * 12 + s] = acc * inv_sqrt_dh;
    }
    __syncthreads();

    if (tid < HALF_TOKENS * H_HEADS) {          // softmax over 9 slots
      float* p = sProb + tid * 12;
      float mx = p[0];
      for (int s = 1; s < 9; ++s) mx = fmaxf(mx, p[s]);
      float sum = 0.f;
      for (int s = 0; s < 9; ++s) { p[s] = __expf(p[s] - mx); sum += p[s]; }
      float inv = 1.f / sum;
      for (int s = 0; s < 9; ++s) p[s] *= inv;
    }
    __syncthreads();

    // 5d: prob-mix raw rows (linearity of V-projection); overlays sK
    for (int e = tid; e < 64 * C_DIM; e += NTHREADS) {
      int h = e / (8 * C_DIM), t = (e / C_DIM) & 7, j = e % C_DIM;
      const float* p = sProb + (t * 8 + h) * 12;
      float acc = p[8] * bf2f(sNow[(size_t)(tb + t) * C_DIM + j]);
#pragma unroll
      for (int s = 0; s < 8; ++s)
        acc += p[s] * bf2f(sGat[((size_t)t * 8 + s) * C_DIM + j]);
      sMix[((size_t)h * 8 + t) * C_DIM + j] = f2bf(acc);
    }
    __syncthreads();

    // 5e: per-head ctx GEMM: ctx[t, h*40 : h*40+40] = mixed[h] @ Wv[:, head slice]
    for (int task = wave; task < 24; task += 8) {
      int h = task / 3, ct = task % 3;
      const unsigned short* arow = sMix + ((size_t)h * 8 + (lane & 15)) * C_DIM;
      int cn = ct * 16 + (lane & 15);                  // 0..47 within head (40 valid)
      int n  = h * DH + ((cn < DH) ? cn : (DH - 1));   // clamp OOB cols, skip their stores
      v8f acc = {};
#pragma unroll
      for (int kt = 0; kt < 10; ++kt)
        acc = wmma_bf16(load_a_frag(arow, kt, lane), load_b_frag(wtv, n, kt, lane), acc);
      if (!(lane & 16) && cn < DH) {                   // only rows 0..7 (= valid tokens)
#pragma unroll
        for (int vi = 0; vi < 8; ++vi)
          sCtx[(size_t)(tb + vi) * C_DIM + h * DH + cn] = f2bf(acc[vi]);
      }
    }
    __syncthreads();
  }

  // Phase 6: out = ctx @ Wo + bo
  for (int ct = wave; ct < 20; ct += 8) {
    const unsigned short* arow = sCtx + (size_t)(lane & 15) * C_DIM;
    int n = ct * 16 + (lane & 15);
    v8f acc = {};
#pragma unroll
    for (int kt = 0; kt < 10; ++kt)
      acc = wmma_bf16(load_a_frag(arow, kt, lane), load_b_frag(wto, n, kt, lane), acc);
    float b = bo[n];
#pragma unroll
    for (int vi = 0; vi < 8; ++vi) {
      int m = (lane & 16) ? (8 + vi) : vi;
      out[(size_t)(tok0 + m) * C_DIM + n] = acc[vi] + b;
    }
  }
}

extern "C" void kernel_launch(void* const* d_in, const int* in_sizes, int n_in,
                              void* d_out, int out_size, void* d_ws, size_t ws_size,
                              hipStream_t stream) {
  const float* hidden = (const float*)d_in[0];
  const float* cached = (const float*)d_in[1];
  const float* Wq = (const float*)d_in[2];
  const float* Wk = (const float*)d_in[3];
  const float* Wv = (const float*)d_in[4];
  const float* Wo = (const float*)d_in[5];
  const float* bo = (const float*)d_in[6];
  unsigned short* wbf = (unsigned short*)d_ws;   // 5 * 102400 bf16 = 1,024,000 B
  float* out = (float*)d_out;

  prep_weights_kernel<<<(W_ELEMS + 255) / 256, 256, 0, stream>>>(Wq, Wk, Wv, Wo, wbf);

  const int n_tokens = 4 * 4096;                         // B * D
  const int blocks = n_tokens / TOKENS_PER_BLOCK;        // 1024
  const size_t shmem = 152064;                           // 152KB -> 2 groups / 320KB WGP
  temporal_attn_kernel<<<blocks, NTHREADS, shmem, stream>>>(hidden, cached, wbf, bo, out);
}
